// ParticleNetGeometric_65773129171008
// MI455X (gfx1250) — matrix operand order repaired and checked by reading, not compiled, not run on hardware
//
#include <hip/hip_runtime.h>

typedef __attribute__((ext_vector_type(16))) _Float16 v16h;
typedef __attribute__((ext_vector_type(8)))  _Float16 v8h;
typedef __attribute__((ext_vector_type(8)))  float    v8f;

#define EPSN 1e-5f

// ---------------- utility kernels ----------------
__global__ void k_zero(unsigned* p, long n) {
  long i = (long)blockIdx.x * blockDim.x + threadIdx.x;
  if (i < n) p[i] = 0u;
}

__global__ void k_cast16(const float* __restrict__ s, _Float16* __restrict__ d, long n) {
  long i = (long)blockIdx.x * blockDim.x + threadIdx.x;
  if (i < n) d[i] = (_Float16)s[i];
}

// Pack f32 [K x H] row-major weight into fp16 WMMA B-fragment order, K padded
// to Kpad with zeros. Fragment layout (per 32x16 K-tile x N-tile):
//   packed[(((nt*KT)+kt)*32 + lane)*16 + e] = W[kt*32 + (e>>3)*16 + (lane>>4)*8 + (e&7)][nt*16 + (lane&15)]
// so each lane's 16 halves are contiguous (32B) -> 2x b128 loads in the GEMM.
__global__ void k_pack_bfrag(const float* __restrict__ s, _Float16* __restrict__ d,
                             int K, int H, int Kpad) {
  int KT = Kpad >> 5;
  long total = (long)KT * (H >> 4) * 512;
  long i = (long)blockIdx.x * blockDim.x + threadIdx.x;
  if (i >= total) return;
  int e  = (int)(i & 15);
  int l  = (int)((i >> 4) & 31);
  int p  = (int)(i >> 9);              // nt*KT + kt
  int kt = p % KT, nt = p / KT;
  int kg = kt * 32 + ((e >> 3) << 4) + ((l >> 4) << 3) + (e & 7);
  int ng = nt * 16 + (l & 15);
  d[i] = (kg < K) ? (_Float16)s[kg * H + ng] : (_Float16)0.0f;
}

__global__ void k_count(const int* __restrict__ batch, int* __restrict__ cnt, int N) {
  int i = blockIdx.x * blockDim.x + threadIdx.x;
  if (i < N) atomicAdd(&cnt[batch[i]], 1);
}

// ---------------- fused EdgeConv MLP (3 stages) + scatter-max ----------------
// One wave = one 32-edge tile (two 16-row A tiles sharing every B fragment,
// halving weight-stream traffic per edge). Stage-0 A fragments gathered
// straight into VGPRs (vectorized v8h chunks when DIN%8==0); inter-stage
// activations stored in next-stage fragment order in per-wave LDS; B
// fragments are single v16h loads from pre-permuted weights. Final stage
// scatters via uint-punned atomic max (valid: post-ReLU values >= 0).
template<int DIN, int H, int KA, int WPB>
__global__ void k_edge_mlp(const _Float16* __restrict__ hin,
                           const int* __restrict__ srcI, const int* __restrict__ dstI,
                           const _Float16* __restrict__ W0, const float* __restrict__ b0,
                           const _Float16* __restrict__ W1, const float* __restrict__ b1,
                           const _Float16* __restrict__ W2, const float* __restrict__ b2,
                           float* __restrict__ agg, int E) {
  constexpr int FR = H / 32;                 // inter-stage K-tiles (>= KA/32)
  __shared__ __align__(32) _Float16 fbuf[WPB][2][FR][32][16];
  __shared__ int sDst[WPB][32];

  const int lane = threadIdx.x & 31;
  const int w    = threadIdx.x >> 5;
  const long tile = (long)blockIdx.x * WPB + w;  // 32-edge tile
  const long e0   = tile * 32;
  if (e0 >= E) return;
  const int r    = lane & 15;
  const int half = lane >> 4;

  {
    long eid = e0 + lane;
    sDst[w][lane] = (eid < E) ? dstI[eid] : 0;
  }

  // Stage-0 A fragments (ISA 16-bit 16x32 layout): lane holds row r; element
  // e -> K = kt*32 + (e>>3)*16 + half*8 + (e&7), i.e. two contiguous 8-runs.
  v16h afrag[2][FR];
#pragma unroll
  for (int t = 0; t < 2; ++t) {
    long eid = e0 + t * 16 + r;
    bool ok = (eid < E);
    int dI = ok ? dstI[eid] : 0;
    int sI = ok ? srcI[eid] : 0;
#pragma unroll
    for (int kt = 0; kt < FR; ++kt) {
      if (kt < KA / 32) {
        v16h av;
        if constexpr ((DIN % 8) == 0) {
#pragma unroll
          for (int g2 = 0; g2 < 2; ++g2) {
            int kbase = kt * 32 + g2 * 16 + half * 8;  // 8 consecutive K
            v8h ch = {};
            if (ok) {
              if (kbase < DIN) {
                ch = *reinterpret_cast<const v8h*>(&hin[dI * DIN + kbase]);
              } else {
                v8h xs = *reinterpret_cast<const v8h*>(&hin[sI * DIN + kbase - DIN]);
                v8h xd = *reinterpret_cast<const v8h*>(&hin[dI * DIN + kbase - DIN]);
                ch = xs - xd;
              }
            }
#pragma unroll
            for (int e2 = 0; e2 < 8; ++e2) av[g2 * 8 + e2] = ch[e2];
          }
        } else {
#pragma unroll
          for (int e = 0; e < 16; ++e) {
            int k = kt * 32 + ((e >> 3) << 4) + half * 8 + (e & 7);
            _Float16 v = (_Float16)0.0f;
            if (k < 2 * DIN && ok) {
              v = (k < DIN) ? hin[dI * DIN + k]
                            : (_Float16)(hin[sI * DIN + (k - DIN)] - hin[dI * DIN + (k - DIN)]);
            }
            av[e] = v;
          }
        }
        afrag[t][kt] = av;
      }
    }
  }

  const _Float16* Ws[3] = {W0, W1, W2};
  const float*    bs[3] = {b0, b1, b2};
#pragma unroll
  for (int s = 0; s < 3; ++s) {
    const int KT = (s == 0) ? (KA / 32) : FR;
    const _Float16* __restrict__ Wt = Ws[s];
    const float*    __restrict__ bt = bs[s];

    if (s > 0) {
      // Reload A fragments written by the previous stage (wave-local DS ops
      // are in-order; fragments land in registers before we overwrite fbuf).
#pragma unroll
      for (int t = 0; t < 2; ++t)
#pragma unroll
        for (int kt = 0; kt < FR; ++kt)
          afrag[t][kt] = *reinterpret_cast<const v16h*>(&fbuf[w][t][kt][lane][0]);
    }

    int dstm[2][8];
    if (s == 2) {
#pragma unroll
      for (int t = 0; t < 2; ++t)
#pragma unroll
        for (int r8 = 0; r8 < 8; ++r8)
          dstm[t][r8] = sDst[w][t * 16 + r8 + half * 8];
    }

    for (int nt = 0; nt < H / 16; ++nt) {
      v8f acc0 = {}, acc1 = {};
#pragma unroll
      for (int kt = 0; kt < FR; ++kt) {
        if (kt < KT) {
          v16h bf = *reinterpret_cast<const v16h*>(
              &Wt[(((long)nt * KT + kt) * 32 + lane) * 16]);
          acc0 = __builtin_amdgcn_wmma_f32_16x16x32_f16(false, afrag[0][kt], false, bf,
                                                        (short)0, acc0, false, false);
          acc1 = __builtin_amdgcn_wmma_f32_16x16x32_f16(false, afrag[1][kt], false, bf,
                                                        (short)0, acc1, false, false);
        }
      }
      float bias = bt[nt * 16 + r];
      if (s < 2) {
        // Write output (m, ncol) in NEXT stage's A-fragment order.
        int ncol = nt * 16 + r;
        int kt2  = ncol >> 5;
        int off  = ncol & 31;
        int e2   = ((off >> 4) << 3) + (off & 7);
        int ln2  = ((off >> 3) & 1) * 16;
#pragma unroll
        for (int r8 = 0; r8 < 8; ++r8) {
          int m = r8 + half * 8;
          fbuf[w][0][kt2][ln2 + m][e2] = (_Float16)fmaxf(acc0[r8] + bias, 0.0f);
          fbuf[w][1][kt2][ln2 + m][e2] = (_Float16)fmaxf(acc1[r8] + bias, 0.0f);
        }
      } else {
#pragma unroll
        for (int r8 = 0; r8 < 8; ++r8) {
          int m = r8 + half * 8;
          float v0 = fmaxf(acc0[r8] + bias, 0.0f);
          float v1 = fmaxf(acc1[r8] + bias, 0.0f);
          if (e0 + m < E)
            atomicMax((unsigned*)&agg[(long)dstm[0][r8] * H + nt * 16 + r],
                      __float_as_uint(v0));
          if (e0 + 16 + m < E)
            atomicMax((unsigned*)&agg[(long)dstm[1][r8] * H + nt * 16 + r],
                      __float_as_uint(v1));
        }
      }
    }
  }
}

// ---------------- node shortcut: agg += x @ Wsc + bsc (WMMA, LDS-free) ------
template<int DIN, int H, int KSC, int WPB>
__global__ void k_shortcut(const _Float16* __restrict__ hin,
                           const _Float16* __restrict__ Wsc, const float* __restrict__ bsc,
                           float* __restrict__ agg, int N) {
  const int lane = threadIdx.x & 31;
  const int w    = threadIdx.x >> 5;
  const int tile = blockIdx.x * WPB + w;
  const int n0   = tile * 16;
  if (n0 >= N) return;
  const int r    = lane & 15;
  const int half = lane >> 4;
  const int node = n0 + r;
  constexpr int KT = KSC / 32;

  v16h afrag[KT];
#pragma unroll
  for (int kt = 0; kt < KT; ++kt) {
    v16h av;
    if constexpr ((DIN % 8) == 0) {   // KSC == DIN: chunks always in-range
#pragma unroll
      for (int g2 = 0; g2 < 2; ++g2) {
        int kbase = kt * 32 + g2 * 16 + half * 8;
        v8h ch = {};
        if (node < N) ch = *reinterpret_cast<const v8h*>(&hin[node * DIN + kbase]);
#pragma unroll
        for (int e2 = 0; e2 < 8; ++e2) av[g2 * 8 + e2] = ch[e2];
      }
    } else {
#pragma unroll
      for (int e = 0; e < 16; ++e) {
        int k = kt * 32 + ((e >> 3) << 4) + half * 8 + (e & 7);
        _Float16 v = (_Float16)0.0f;
        if (k < DIN && node < N) v = hin[node * DIN + k];
        av[e] = v;
      }
    }
    afrag[kt] = av;
  }

  for (int nt = 0; nt < H / 16; ++nt) {
    v8f acc = {};
#pragma unroll
    for (int kt = 0; kt < KT; ++kt) {
      v16h bf = *reinterpret_cast<const v16h*>(
          &Wsc[(((long)nt * KT + kt) * 32 + lane) * 16]);
      acc = __builtin_amdgcn_wmma_f32_16x16x32_f16(false, afrag[kt], false, bf,
                                                   (short)0, acc, false, false);
    }
    float bias = bsc[nt * 16 + r];
#pragma unroll
    for (int r8 = 0; r8 < 8; ++r8) {
      int nd = n0 + r8 + half * 8;
      if (nd < N) {
        long o = (long)nd * H + nt * 16 + r;
        agg[o] = agg[o] + acc[r8] + bias;
      }
    }
  }
}

// ---------------- instance-norm stats / normalize / pooling ----------------
template<int H>
__global__ void k_stats(const float* __restrict__ pre, const int* __restrict__ batch,
                        float* __restrict__ sum, float* __restrict__ sumsq, int N) {
  long i = (long)blockIdx.x * blockDim.x + threadIdx.x;
  if (i >= (long)N * H) return;
  int node = (int)(i / H), c = (int)(i - (long)node * H);
  int g = batch[node];
  float v = pre[i];
  atomicAdd(&sum[g * H + c], v);
  atomicAdd(&sumsq[g * H + c], v * v);
}

template<int H>
__global__ void k_norm(const float* __restrict__ pre, const int* __restrict__ batch,
                       const float* __restrict__ sum, const float* __restrict__ sumsq,
                       const int* __restrict__ cnt, _Float16* __restrict__ out, int N) {
  long i = (long)blockIdx.x * blockDim.x + threadIdx.x;
  if (i >= (long)N * H) return;
  int node = (int)(i / H), c = (int)(i - (long)node * H);
  int g = batch[node];
  float cn   = fmaxf((float)cnt[g], 1.0f);
  float mean = sum[g * H + c] / cn;
  float var  = sumsq[g * H + c] / cn - mean * mean;
  float y    = (pre[i] - mean) * rsqrtf(var + EPSN);
  out[i] = (_Float16)fmaxf(y, 0.0f);
}

template<int H>
__global__ void k_pool(const _Float16* __restrict__ h, const int* __restrict__ batch,
                       float* __restrict__ psum, float* __restrict__ pmax, int N) {
  long i = (long)blockIdx.x * blockDim.x + threadIdx.x;
  if (i >= (long)N * H) return;
  int node = (int)(i / H), c = (int)(i - (long)node * H);
  int g = batch[node];
  float v = (float)h[i];
  atomicAdd(&psum[g * H + c], v);
  atomicMax((unsigned*)&pmax[g * H + c], __float_as_uint(v));  // v >= 0
}

// ---------------- head MLP: [G,512] -> 256 -> 2 -> out1-out0 ----------------
__global__ void k_head(const float* __restrict__ psum, const float* __restrict__ pmax,
                       const int* __restrict__ cnt,
                       const float* __restrict__ W0, const float* __restrict__ b0,
                       const float* __restrict__ W1, const float* __restrict__ b1,
                       float* __restrict__ out, int G) {
  __shared__ float p[512];
  __shared__ float q[256];
  int g = blockIdx.x;
  int t = threadIdx.x;     // 256 threads
  if (g >= G) return;
  float cn = fmaxf((float)cnt[g], 1.0f);
  p[t]       = psum[g * 256 + t] / cn;
  p[256 + t] = pmax[g * 256 + t];
  __syncthreads();
  float acc = b0[t];
  for (int k = 0; k < 512; ++k) acc += p[k] * W0[k * 256 + t];
  q[t] = fmaxf(acc, 0.0f);
  __syncthreads();
  if (t == 0) {
    float o0 = b1[0], o1 = b1[1];
    for (int k = 0; k < 256; ++k) { o0 += q[k] * W1[k * 2]; o1 += q[k] * W1[k * 2 + 1]; }
    out[g] = o1 - o0;
  }
}

// ---------------- host orchestration ----------------
static inline long cdivl(long a, long b) { return (a + b - 1) / b; }

extern "C" void kernel_launch(void* const* d_in, const int* in_sizes, int n_in,
                              void* d_out, int out_size, void* d_ws, size_t ws_size,
                              hipStream_t stream) {
  const float* x     = (const float*)d_in[0];
  const int*   ei    = (const int*)d_in[1];
  const int*   batch = (const int*)d_in[2];
  const int N = in_sizes[0] / 5;
  const int E = in_sizes[1] / 2;
  const int G = out_size;
  const int* srcI = ei;
  const int* dstI = ei + E;

  const float* P[28];
  for (int i = 0; i < 28 && (3 + i) < n_in; ++i) P[i] = (const float*)d_in[3 + i];
  // conv li base = li*8: W0,b0,W1,b1,W2,b2,Wsc,bsc ; fc0_W=P[24],fc0_b=P[25],fc1_W=P[26],fc1_b=P[27]

  // ---- carve workspace ----
  char* ws = (char*)d_ws;
  size_t off = 0;
  auto take = [&](size_t bytes) -> char* {
    char* p = ws + off;
    off += (bytes + 255) & ~(size_t)255;
    return p;
  };
  _Float16* hx   = (_Float16*)take((size_t)N * 5 * 2);
  _Float16* h64  = (_Float16*)take((size_t)N * 64 * 2);
  _Float16* h128 = (_Float16*)take((size_t)N * 128 * 2);
  _Float16* h256 = (_Float16*)take((size_t)N * 256 * 2);
  float*    agg  = (float*)take((size_t)N * 256 * 4);
  float*    sum  = (float*)take((size_t)G * 256 * 4);
  float*    ssq  = (float*)take((size_t)G * 256 * 4);
  float*    psum = (float*)take((size_t)G * 256 * 4);
  float*    pmax = (float*)take((size_t)G * 256 * 4);
  int*      cnt  = (int*)take((size_t)G * 4);
  _Float16* w00  = (_Float16*)take((size_t)32 * 64 * 2);
  _Float16* w01  = (_Float16*)take((size_t)64 * 64 * 2);
  _Float16* w02  = (_Float16*)take((size_t)64 * 64 * 2);
  _Float16* w0s  = (_Float16*)take((size_t)32 * 64 * 2);
  _Float16* w10  = (_Float16*)take((size_t)128 * 128 * 2);
  _Float16* w11  = (_Float16*)take((size_t)128 * 128 * 2);
  _Float16* w12  = (_Float16*)take((size_t)128 * 128 * 2);
  _Float16* w1s  = (_Float16*)take((size_t)64 * 128 * 2);
  _Float16* w20  = (_Float16*)take((size_t)256 * 256 * 2);
  _Float16* w21  = (_Float16*)take((size_t)256 * 256 * 2);
  _Float16* w22  = (_Float16*)take((size_t)256 * 256 * 2);
  _Float16* w2s  = (_Float16*)take((size_t)128 * 256 * 2);
  (void)ws_size;

  const int TPB = 256;
  auto gz = [&](long n) { return (unsigned)cdivl(n, TPB); };

  // graph counts
  k_zero<<<gz(G), TPB, 0, stream>>>((unsigned*)cnt, G);
  k_count<<<gz(N), TPB, 0, stream>>>(batch, cnt, N);

  // fp16 inputs + fragment-packed fp16 weights
  k_cast16<<<gz((long)N * 5), TPB, 0, stream>>>(x, hx, (long)N * 5);
  k_pack_bfrag<<<gz(32 * 64),   TPB, 0, stream>>>(P[0],  w00, 10,  64,  32);
  k_pack_bfrag<<<gz(64 * 64),   TPB, 0, stream>>>(P[2],  w01, 64,  64,  64);
  k_pack_bfrag<<<gz(64 * 64),   TPB, 0, stream>>>(P[4],  w02, 64,  64,  64);
  k_pack_bfrag<<<gz(32 * 64),   TPB, 0, stream>>>(P[6],  w0s, 5,   64,  32);
  k_pack_bfrag<<<gz(128 * 128), TPB, 0, stream>>>(P[8],  w10, 128, 128, 128);
  k_pack_bfrag<<<gz(128 * 128), TPB, 0, stream>>>(P[10], w11, 128, 128, 128);
  k_pack_bfrag<<<gz(128 * 128), TPB, 0, stream>>>(P[12], w12, 128, 128, 128);
  k_pack_bfrag<<<gz(64 * 128),  TPB, 0, stream>>>(P[14], w1s, 64,  128, 64);
  k_pack_bfrag<<<gz(256 * 256), TPB, 0, stream>>>(P[16], w20, 256, 256, 256);
  k_pack_bfrag<<<gz(256 * 256), TPB, 0, stream>>>(P[18], w21, 256, 256, 256);
  k_pack_bfrag<<<gz(256 * 256), TPB, 0, stream>>>(P[20], w22, 256, 256, 256);
  k_pack_bfrag<<<gz(128 * 256), TPB, 0, stream>>>(P[22], w2s, 128, 256, 128);

  const long tilesE32 = cdivl(E, 32);
  const long tilesN   = cdivl(N, 16);

  // ---- layer 0: din=5 (KA=32), h=64 ----
  k_zero<<<gz((long)N * 64), TPB, 0, stream>>>((unsigned*)agg, (long)N * 64);
  k_edge_mlp<5, 64, 32, 4><<<(unsigned)cdivl(tilesE32, 4), 4 * 32, 0, stream>>>(
      hx, srcI, dstI, w00, P[1], w01, P[3], w02, P[5], agg, E);
  k_shortcut<5, 64, 32, 8><<<(unsigned)cdivl(tilesN, 8), 8 * 32, 0, stream>>>(
      hx, w0s, P[7], agg, N);
  k_zero<<<gz((long)G * 64), TPB, 0, stream>>>((unsigned*)sum, (long)G * 64);
  k_zero<<<gz((long)G * 64), TPB, 0, stream>>>((unsigned*)ssq, (long)G * 64);
  k_stats<64><<<gz((long)N * 64), TPB, 0, stream>>>(agg, batch, sum, ssq, N);
  k_norm<64><<<gz((long)N * 64), TPB, 0, stream>>>(agg, batch, sum, ssq, cnt, h64, N);

  // ---- layer 1: din=64 (KA=128), h=128 ----
  k_zero<<<gz((long)N * 128), TPB, 0, stream>>>((unsigned*)agg, (long)N * 128);
  k_edge_mlp<64, 128, 128, 4><<<(unsigned)cdivl(tilesE32, 4), 4 * 32, 0, stream>>>(
      h64, srcI, dstI, w10, P[9], w11, P[11], w12, P[13], agg, E);
  k_shortcut<64, 128, 64, 8><<<(unsigned)cdivl(tilesN, 8), 8 * 32, 0, stream>>>(
      h64, w1s, P[15], agg, N);
  k_zero<<<gz((long)G * 128), TPB, 0, stream>>>((unsigned*)sum, (long)G * 128);
  k_zero<<<gz((long)G * 128), TPB, 0, stream>>>((unsigned*)ssq, (long)G * 128);
  k_stats<128><<<gz((long)N * 128), TPB, 0, stream>>>(agg, batch, sum, ssq, N);
  k_norm<128><<<gz((long)N * 128), TPB, 0, stream>>>(agg, batch, sum, ssq, cnt, h128, N);

  // ---- layer 2: din=128 (KA=256), h=256 ----
  k_zero<<<gz((long)N * 256), TPB, 0, stream>>>((unsigned*)agg, (long)N * 256);
  k_edge_mlp<128, 256, 256, 2><<<(unsigned)cdivl(tilesE32, 2), 2 * 32, 0, stream>>>(
      h128, srcI, dstI, w20, P[17], w21, P[19], w22, P[21], agg, E);
  k_shortcut<128, 256, 128, 8><<<(unsigned)cdivl(tilesN, 8), 8 * 32, 0, stream>>>(
      h128, w2s, P[23], agg, N);
  k_zero<<<gz((long)G * 256), TPB, 0, stream>>>((unsigned*)sum, (long)G * 256);
  k_zero<<<gz((long)G * 256), TPB, 0, stream>>>((unsigned*)ssq, (long)G * 256);
  k_stats<256><<<gz((long)N * 256), TPB, 0, stream>>>(agg, batch, sum, ssq, N);
  k_norm<256><<<gz((long)N * 256), TPB, 0, stream>>>(agg, batch, sum, ssq, cnt, h256, N);

  // ---- mean/max pooling + head MLP ----
  k_zero<<<gz((long)G * 256), TPB, 0, stream>>>((unsigned*)psum, (long)G * 256);
  k_zero<<<gz((long)G * 256), TPB, 0, stream>>>((unsigned*)pmax, (long)G * 256);
  k_pool<256><<<gz((long)N * 256), TPB, 0, stream>>>(h256, batch, psum, pmax, N);
  k_head<<<(unsigned)G, 256, 0, stream>>>(psum, pmax, cnt, P[24], P[25], P[26], P[27],
                                          (float*)d_out, G);
}
